// Projection_Attention_69243462746806
// MI455X (gfx1250) — matrix operand order repaired and checked by reading, not compiled, not run on hardware
//
#include <hip/hip_runtime.h>
#include <hip/hip_bf16.h>
#include <stdint.h>

// ---------------------------------------------------------------------------
// Problem constants (reference: B=4, S=4096, D=1024, H=16, HID=ATT=64)
// ---------------------------------------------------------------------------
#define NTOK      16384          // B*S tokens
#define DMODEL    1024           // K dimension of projections
#define NOUT      3072           // q|k|v concatenated output features
#define NT_TOTAL  192            // NOUT / 16 N-tiles
#define KSTEPS    32             // DMODEL / 32 K-steps (bf16 WMMA K=32)
#define WAVES     16             // waves per workgroup (512 threads)
#define NT_PER_WAVE (NT_TOTAL / WAVES)   // 12
#define NGROUPS   1024           // NTOK / 16 token groups
#define FRAG_SHORTS (32 * 16)    // one 16x32 / 32x16 bf16 tile: 32 lanes x 16 elems

typedef __bf16 v16bf __attribute__((ext_vector_type(16)));
typedef float  v8f   __attribute__((ext_vector_type(8)));

#if __has_builtin(__builtin_amdgcn_sched_barrier)
#define SCHED_FENCE() __builtin_amdgcn_sched_barrier(0)
#else
#define SCHED_FENCE()
#endif

__device__ __forceinline__ unsigned short f32_to_bf16_rne(float f) {
  unsigned int u = __float_as_uint(f);
  unsigned int r = u + 0x7FFFu + ((u >> 16) & 1u);   // round-to-nearest-even
  return (unsigned short)(r >> 16);
}

// ---------------------------------------------------------------------------
// Pack Wq|Wk|Wv (row-major KxN fp32) into B-fragment-major bf16:
//   Wpack[((nt*KSTEPS + ks)*32 + lane)*16 + e]
// B 32x16 bf16 layout: lane = {n = lane%16, half = lane/16}, elem e -> K = (e<8 ? e : 8+e) + 8*half
// ---------------------------------------------------------------------------
__global__ void __launch_bounds__(256)
pack_w_kernel(const float* __restrict__ Wq,
              const float* __restrict__ Wk,
              const float* __restrict__ Wv,
              unsigned short* __restrict__ Wpack) {
  int idx = blockIdx.x * blockDim.x + threadIdx.x;       // one per (nt,ks,lane)
  if (idx >= NT_TOTAL * KSTEPS * 32) return;
  int lane = idx & 31;
  int ks   = (idx >> 5) & 31;
  int nt   = idx >> 10;
  int mat  = nt >> 6;                                    // 0=q 1=k 2=v
  const float* W = (mat == 0) ? Wq : (mat == 1) ? Wk : Wv;
  int ncol = ((nt & 63) << 4) + (lane & 15);
  int half = lane >> 4;
  unsigned short frag[16] __attribute__((aligned(16)));
#pragma unroll
  for (int e = 0; e < 16; ++e) {
    int K = ks * 32 + ((e < 8) ? e : (8 + e)) + 8 * half;
    frag[e] = f32_to_bf16_rne(W[(size_t)K * 1024 + ncol]);
  }
  uint4* dst = (uint4*)(Wpack + (size_t)idx * 16);
  dst[0] = *(const uint4*)&frag[0];
  dst[1] = *(const uint4*)&frag[8];
}

// ---------------------------------------------------------------------------
// Pack x (token-major [16384][1024] fp32) into A-fragment-major bf16:
//   Xpack[((tg*KSTEPS + ks)*32 + lane)*16 + e]
// A 16x32 bf16 layout (ISA table): lane = {m = lane%16, half = lane/16},
//   elem e -> reg r=e/2, pos=e%2, K = (r<4 ? 2r : 16+2(r-4)) + 8*half + pos
// ---------------------------------------------------------------------------
__global__ void __launch_bounds__(256)
pack_x_kernel(const float* __restrict__ x,
              unsigned short* __restrict__ Xpack) {
  int idx = blockIdx.x * blockDim.x + threadIdx.x;       // one per (tg,ks,lane)
  if (idx >= NGROUPS * KSTEPS * 32) return;
  int lane = idx & 31;
  int ks   = (idx >> 5) & 31;
  int tg   = idx >> 10;
  int m    = lane & 15;
  int half = lane >> 4;
  unsigned short frag[16] __attribute__((aligned(16)));
#pragma unroll
  for (int e = 0; e < 16; ++e) {
    int r = e >> 1, pos = e & 1;
    int kb = (r < 4) ? (2 * r) : (16 + 2 * (r - 4));
    int K = ks * 32 + kb + 8 * half + pos;
    frag[e] = f32_to_bf16_rne(x[((size_t)tg * 16 + m) * DMODEL + K]);
  }
  uint4* dst = (uint4*)(Xpack + (size_t)idx * 16);
  dst[0] = *(const uint4*)&frag[0];
  dst[1] = *(const uint4*)&frag[8];
}

// ---------------------------------------------------------------------------
// Fused QKV projection (bf16 WMMA) + per-token 16-head attention.
// Grid: NGROUPS workgroups x 512 threads (16 wave32). Dynamic LDS: 16*3072 fp32.
// __launch_bounds__(512): 4 waves/SIMD -> 256-VGPR budget, so the staged
// fragments (96 accum + 96 B-stage + 16 A ~= 220 VGPRs) fit WITHOUT spilling.
// Inner loop: issue all fragment loads for this K-step (plus next-A prefetch),
// sched_barrier pins them above the 12 WMMAs -> partial s_wait_loadcnt values,
// WMMA issue overlaps the in-flight load stream.
// ---------------------------------------------------------------------------
__global__ void __launch_bounds__(WAVES * 32)
fused_qkv_attn_kernel(const unsigned short* __restrict__ Xpack,
                      const unsigned short* __restrict__ Wpack,
                      const float* __restrict__ bq,
                      const float* __restrict__ bk,
                      const float* __restrict__ bv,
                      float* __restrict__ out) {
  extern __shared__ float lds[];                  // [16 tokens][3072 features]
  const int tg   = blockIdx.x;
  const int tid  = threadIdx.x;
  const int wave = tid >> 5;
  const int lane = tid & 31;

  // -------- phase 1: projection GEMM, M=16 tokens, this wave owns 12 N-tiles
  const v8f vzero = {0.f, 0.f, 0.f, 0.f, 0.f, 0.f, 0.f, 0.f};
  v8f acc[NT_PER_WAVE];
#pragma unroll
  for (int j = 0; j < NT_PER_WAVE; ++j) acc[j] = vzero;

  const unsigned short* xbase =
      Xpack + (((size_t)tg * KSTEPS) * 32 + lane) * 16;
  const unsigned short* wbase =
      Wpack + (((size_t)(wave * NT_PER_WAVE) * KSTEPS) * 32 + lane) * 16;

  v16bf acur = *reinterpret_cast<const v16bf*>(xbase);   // A for ks = 0

#pragma unroll 1
  for (int ks = 0; ks < KSTEPS; ++ks) {
    // ---- load stage: 12 B fragments + next A fragment, all issued up front
    v16bf bst[NT_PER_WAVE];
#pragma unroll
    for (int j = 0; j < NT_PER_WAVE; ++j) {
      bst[j] = *reinterpret_cast<const v16bf*>(
          wbase + ((size_t)j * KSTEPS + ks) * FRAG_SHORTS);
    }
    int ksn = (ks + 1 < KSTEPS) ? (ks + 1) : 0;          // tail re-reads ks=0
    v16bf anxt = *reinterpret_cast<const v16bf*>(
        xbase + (size_t)ksn * FRAG_SHORTS);

    SCHED_FENCE();   // keep all loads above, all WMMAs below

    // ---- math stage: 12 WMMAs; each waits only for its own (older) loads
#pragma unroll
    for (int j = 0; j < NT_PER_WAVE; ++j) {
      acc[j] = __builtin_amdgcn_wmma_f32_16x16x32_bf16(
          false, acur, false, bst[j], (short)0, acc[j], false, false);
    }

    SCHED_FENCE();
    acur = anxt;
  }

  // -------- epilogue: bias add, scatter C tiles into LDS (lds[token][feature])
  const int n = lane & 15, half = lane >> 4;
#pragma unroll
  for (int j = 0; j < NT_PER_WAVE; ++j) {
    int nt  = wave * NT_PER_WAVE + j;
    int col = nt * 16 + n;                               // 0..3071
    const float* bias = (nt < 64) ? bq : (nt < 128) ? bk : bv;
    float bval = bias[col & 1023];
#pragma unroll
    for (int r = 0; r < 8; ++r) {
      int m = r + 8 * half;                              // C-layout: row m
      lds[m * NOUT + col] = acc[j][r] + bval;
    }
  }
  __syncthreads();

  // -------- phase 2: per-token head attention (2% of FLOPs -> fp32 VALU)
  if (tid < 256) {
    const int t = tid >> 4;                              // token in group
    const int i = tid & 15;                              // query head
    const float* base = &lds[t * NOUT];

    float qr[64];
#pragma unroll
    for (int d = 0; d < 64; d += 4) {
      float4 v4 = *(const float4*)&base[i * 64 + d];
      qr[d] = v4.x; qr[d + 1] = v4.y; qr[d + 2] = v4.z; qr[d + 3] = v4.w;
    }

    float s[16];
#pragma unroll
    for (int j = 0; j < 16; ++j) {
      const float* kr = &base[1024 + j * 64];            // broadcast reads
      float a = 0.f;
#pragma unroll 8
      for (int d = 0; d < 64; ++d) a = fmaf(qr[d], kr[d], a);
      s[j] = a * 0.125f;                                 // 1/sqrt(64)
    }
    float mx = s[0];
#pragma unroll
    for (int j = 1; j < 16; ++j) mx = fmaxf(mx, s[j]);
    float sum = 0.f;
#pragma unroll
    for (int j = 0; j < 16; ++j) { s[j] = __expf(s[j] - mx); sum += s[j]; }
    const float inv = __frcp_rn(sum);

    float ctx[64];
#pragma unroll
    for (int a = 0; a < 64; ++a) ctx[a] = 0.f;
#pragma unroll 1
    for (int j = 0; j < 16; ++j) {
      float p = s[j] * inv;
      const float* vr = &base[2048 + j * 64];            // broadcast reads
#pragma unroll
      for (int a = 0; a < 64; ++a) ctx[a] = fmaf(p, vr[a], ctx[a]);
    }

    float* op = out + ((size_t)tg * 16 + t) * 1024 + i * 64;
#pragma unroll
    for (int a = 0; a < 64; a += 4) {
      float4 v4 = make_float4(ctx[a], ctx[a + 1], ctx[a + 2], ctx[a + 3]);
      *(float4*)&op[a] = v4;
    }
  }
}

// ---------------------------------------------------------------------------
extern "C" void kernel_launch(void* const* d_in, const int* in_sizes, int n_in,
                              void* d_out, int out_size, void* d_ws, size_t ws_size,
                              hipStream_t stream) {
  const float* x  = (const float*)d_in[0];
  const float* Wq = (const float*)d_in[1];
  const float* bq = (const float*)d_in[2];
  const float* Wk = (const float*)d_in[3];
  const float* bk = (const float*)d_in[4];
  const float* Wv = (const float*)d_in[5];
  const float* bv = (const float*)d_in[6];
  float* out = (float*)d_out;

  // workspace: packed bf16 weights (6 MB) then packed bf16 activations (32 MB)
  unsigned short* Wpack = (unsigned short*)d_ws;
  unsigned short* Xpack = Wpack + (size_t)NT_TOTAL * KSTEPS * 32 * 16;

  {
    int total = NT_TOTAL * KSTEPS * 32;                  // 196608
    pack_w_kernel<<<(total + 255) / 256, 256, 0, stream>>>(Wq, Wk, Wv, Wpack);
  }
  {
    int total = NGROUPS * KSTEPS * 32;                   // 1048576
    pack_x_kernel<<<(total + 255) / 256, 256, 0, stream>>>(x, Xpack);
  }
  {
    size_t ldsBytes = (size_t)16 * NOUT * sizeof(float); // 192 KB of 320 KB WGP LDS
    fused_qkv_attn_kernel<<<NGROUPS, WAVES * 32, ldsBytes, stream>>>(
        Xpack, Wpack, bq, bk, bv, out);
  }
}
